// DeepTransitionRNN_9715216024077
// MI455X (gfx1250) — compile-verified, matrix operand order checked by reading, not run on hardware
//
#include <hip/hip_runtime.h>

// ---------------------------------------------------------------------------
// DeepTransitionRNN for MI455X (gfx1250, wave32, WMMA).
//
// T=512, B=128, D=H=256, L=4.  Batch rows are independent through the
// recurrence, so grid = 8 workgroups x 16 batch rows (M=16 == WMMA tile
// height).  512 threads = 16 waves; wave w owns N-tile w (cols 16w..16w+15)
// of every GEMM.  All GEMMs are [16 x K x 256] with K in {512, 256} and run
// on v_wmma_f32_16x16x32_bf16 (bf16 operands, f32 accumulate).  Weights are
// pre-swizzled into B-fragment-linear bf16 in d_ws (L2-resident, re-streamed
// each timestep).  Hidden state / gates / LayerNorm stay f32 in LDS.
// ---------------------------------------------------------------------------

typedef __attribute__((ext_vector_type(16))) __bf16 v16bf;
typedef __attribute__((ext_vector_type(8)))  __bf16 v8bf;
typedef __attribute__((ext_vector_type(8)))  float  v8f;

#define T_STEPS 512
#define BATCH   128
#define DIM     256
#define HID     256
#define NLAYER  4
#define ROWS    16
#define CATS    528   // padded bf16 row stride of the [x|h] LDS tile (16B aligned)

union FragU  { v16bf v; v8bf h[2]; };
union PackBF { v8bf v; unsigned short s[8]; };

__device__ __forceinline__ unsigned short f2bf(float f) {
  // native f32 -> bf16 converter (RTNE) instead of a 3-op integer bit trick
  return __builtin_bit_cast(unsigned short, (__bf16)f);
}

__device__ __forceinline__ float fast_sigmoid(float v) {
  return __builtin_amdgcn_rcpf(1.f + __expf(-v));
}
__device__ __forceinline__ float fast_tanh(float v) {
  // tanh(x) = 1 - 2/(e^{2x}+1); correct limits at +/-inf
  return 1.f - 2.f * __builtin_amdgcn_rcpf(__expf(2.f * v) + 1.f);
}

// A-fragment (16x32 bf16, M x K) from LDS.  lane<16 holds K {k0..k0+7,
// k0+16..k0+23} of row m=lane&15; lanes 16-31 hold the other K-halves.
__device__ __forceinline__ v16bf lds_a_frag(const unsigned short* __restrict__ base,
                                            int m, int kt, int colOff, int laneHalf) {
  const int k0 = colOff + kt * 32 + laneHalf * 8;
  FragU u;
  u.h[0] = *(const v8bf*)(base + m * CATS + k0);
  u.h[1] = *(const v8bf*)(base + m * CATS + k0 + 16);
  return u.v;
}

// B-fragment (32x16 bf16, K x N) from pre-swizzled global weights: each
// lane's 16 values are contiguous (32 bytes -> two global_load_b128).
__device__ __forceinline__ v16bf g_b_frag(const unsigned short* __restrict__ w,
                                          int nt, int kt, int KT, int lane) {
  const unsigned short* p = w + (((size_t)(nt * KT + kt) * 32 + lane) << 4);
  FragU u;
  u.h[0] = *(const v8bf*)(p);
  u.h[1] = *(const v8bf*)(p + 8);
  return u.v;
}

#define WMMA_BF16(A, B, C) \
  __builtin_amdgcn_wmma_f32_16x16x32_bf16(false, (A), false, (B), (short)0, (C), false, false)

// LayerNorm + sigmoid over one row (256 f32 in LDS), one wave per row.
__device__ __forceinline__ void ln_sigmoid_row(float* __restrict__ buf, int row, int lane,
                                               const float* __restrict__ g,
                                               const float* __restrict__ b) {
  float* p = buf + row * HID + lane * 8;
  float vals[8];
  float s = 0.f, ss = 0.f;
#pragma unroll
  for (int j = 0; j < 8; ++j) { float v = p[j]; vals[j] = v; s += v; ss += v * v; }
#pragma unroll
  for (int off = 16; off > 0; off >>= 1) {
    s  += __shfl_xor(s,  off, 32);
    ss += __shfl_xor(ss, off, 32);
  }
  const float mean = s * (1.f / 256.f);
  const float var  = ss * (1.f / 256.f) - mean * mean;
  const float rstd = __builtin_amdgcn_rsqf(var + 1e-5f);
  const int n0 = lane * 8;
#pragma unroll
  for (int j = 0; j < 8; ++j) {
    float t = (vals[j] - mean) * rstd * g[n0 + j] + b[n0 + j];
    p[j] = fast_sigmoid(t);
  }
}

// --------------------------- weight pre-swizzle ----------------------------
// Maps W[K,N] (row-major f32) into WMMA B-fragment-linear bf16:
// dst[((nt*KT+kt)*32+lane)*16+j] = W[kt*32+klocal(lane,j)][nt*16+(lane&15)]
__global__ void swizzle_w(const float* __restrict__ src,
                          unsigned short* __restrict__ dst, int K, int N) {
  const int total = K * N;
  const int f = blockIdx.x * blockDim.x + threadIdx.x;
  if (f >= total) return;
  const int j        = f & 15;
  const int lane     = (f >> 4) & 31;
  const int tile     = f >> 9;
  const int KT       = K >> 5;
  const int kt       = tile % KT;
  const int nt       = tile / KT;
  const int laneHalf = lane >> 4;
  const int v        = j >> 1;
  const int klocal   = ((v >> 2) << 4) + (laneHalf << 3) + ((v & 3) << 1) + (j & 1);
  const int k        = kt * 32 + klocal;
  const int n        = nt * 16 + (lane & 15);
  dst[f] = f2bf(src[k * N + n]);
}

// ------------------------------ main kernel --------------------------------
__global__ __launch_bounds__(512, 1)
void rnn_kernel(const float* __restrict__ x, const int* __restrict__ lengths,
                const unsigned short* __restrict__ wWr, const unsigned short* __restrict__ wWz,
                const unsigned short* __restrict__ wWl, const unsigned short* __restrict__ wCx,
                const unsigned short* __restrict__ wCh, const unsigned short* __restrict__ wWlt,
                const unsigned short* __restrict__ wtWr, const unsigned short* __restrict__ wtWz,
                const unsigned short* __restrict__ wtWc,
                const float* __restrict__ g_r, const float* __restrict__ b_r,
                const float* __restrict__ g_z, const float* __restrict__ b_z,
                const float* __restrict__ g_l, const float* __restrict__ b_l,
                const float* __restrict__ chb,
                const float* __restrict__ tg_r, const float* __restrict__ tb_r,
                const float* __restrict__ tg_z, const float* __restrict__ tb_z,
                const float* __restrict__ tb_c,
                float* __restrict__ out) {
  extern __shared__ unsigned char smem[];
  unsigned short* cat = (unsigned short*)smem;          // [16][CATS] bf16: [x | h]
  float* hbuf  = (float*)(smem + ROWS * CATS * 2);      // [16][256]
  float* hprev = hbuf  + ROWS * HID;
  float* rb    = hprev + ROWS * HID;
  float* zb    = rb    + ROWS * HID;
  float* lb    = zb    + ROWS * HID;
  float* S0    = lb    + ROWS * HID;
  float* S1    = S0    + ROWS * HID;
  float* S2    = S1    + ROWS * HID;

  const int tid   = threadIdx.x;
  const int wave  = tid >> 5;        // 0..15: N-tile for GEMMs, row for LN
  const int lane  = tid & 31;
  const int lh    = lane >> 4;       // lane half for fragment layout
  const int am    = lane & 15;       // A-fragment row
  const int b0    = blockIdx.x * ROWS;
  const int mrow  = tid >> 5;        // row handled in elementwise phases
  const int ncol0 = (tid & 31) * 8;  // 8 contiguous cols per thread
  const int mylen = lengths[b0 + mrow];

  // init h = 0
  {
    PackBF zb16; 
#pragma unroll
    for (int j = 0; j < 8; ++j) zb16.s[j] = 0;
#pragma unroll
    for (int j = 0; j < 8; ++j) {
      const int idx = mrow * HID + ncol0 + j;
      hbuf[idx] = 0.f; hprev[idx] = 0.f;
    }
    *(v8bf*)(&cat[mrow * CATS + DIM + ncol0]) = zb16.v;
  }
  __syncthreads();

  for (int t = 0; t < T_STEPS; ++t) {
    // ---- stage x[t] tile as bf16 into cat[:, 0:256) (one b128 DS store) ----
    {
      const float* xp = &x[(((size_t)t * BATCH) + b0 + mrow) * DIM + ncol0];
      PackBF xb;
#pragma unroll
      for (int j = 0; j < 8; ++j) xb.s[j] = f2bf(xp[j]);
      *(v8bf*)(&cat[mrow * CATS + ncol0]) = xb.v;
    }
    __syncthreads();

    // ---- gate GEMMs: cat @ {Wr,Wz,Wl}  (K = 512) ----
    {
      v8f aR = {}, aZ = {}, aL = {};
#pragma unroll 4
      for (int kt = 0; kt < 16; ++kt) {
        v16bf a  = lds_a_frag(cat, am, kt, 0, lh);
        v16bf br = g_b_frag(wWr, wave, kt, 16, lane);
        v16bf bz = g_b_frag(wWz, wave, kt, 16, lane);
        v16bf bl = g_b_frag(wWl, wave, kt, 16, lane);
        aR = WMMA_BF16(a, br, aR);
        aZ = WMMA_BF16(a, bz, aZ);
        aL = WMMA_BF16(a, bl, aL);
      }
      const int n  = wave * 16 + am;
      const int mb = lh * 8;                   // C/D layout: VGPR i -> row mb+i
#pragma unroll
      for (int i = 0; i < 8; ++i) {
        rb[(mb + i) * HID + n] = aR[i];
        zb[(mb + i) * HID + n] = aZ[i];
        lb[(mb + i) * HID + n] = aL[i];
      }
    }
    __syncthreads();

    // ---- LayerNorm + sigmoid for r, z, l (one wave per row) ----
    ln_sigmoid_row(rb, wave, lane, g_r, b_r);
    ln_sigmoid_row(zb, wave, lane, g_z, b_z);
    ln_sigmoid_row(lb, wave, lane, g_l, b_l);
    __syncthreads();

    // ---- x@Cx -> S0, h@Ch_w -> S1, x@Wlt -> S2  (K = 256) ----
    {
      v8f a0 = {}, a1 = {}, a2 = {};
#pragma unroll 2
      for (int kt = 0; kt < 8; ++kt) {
        v16bf ax = lds_a_frag(cat, am, kt, 0, lh);
        v16bf ah = lds_a_frag(cat, am, kt, DIM, lh);
        a0 = WMMA_BF16(ax, g_b_frag(wCx,  wave, kt, 8, lane), a0);
        a1 = WMMA_BF16(ah, g_b_frag(wCh,  wave, kt, 8, lane), a1);
        a2 = WMMA_BF16(ax, g_b_frag(wWlt, wave, kt, 8, lane), a2);
      }
      const int n  = wave * 16 + am;
      const int mb = lh * 8;
#pragma unroll
      for (int i = 0; i < 8; ++i) {
        S0[(mb + i) * HID + n] = a0[i];
        S1[(mb + i) * HID + n] = a1[i];
        S2[(mb + i) * HID + n] = a2[i];
      }
    }
    __syncthreads();

    // ---- n = tanh(xCx + r*(hCh + Ch_b)) + l*xWlt ;  h = (1-z)h + z*n ----
    {
      PackBF hb;
#pragma unroll
      for (int j = 0; j < 8; ++j) {
        const int n   = ncol0 + j;
        const int idx = mrow * HID + n;
        const float hch = S1[idx] + chb[n];
        const float nn  = fast_tanh(S0[idx] + rb[idx] * hch) + lb[idx] * S2[idx];
        const float z   = zb[idx];
        const float hv  = (1.f - z) * hbuf[idx] + z * nn;
        hbuf[idx] = hv;
        hb.s[j] = f2bf(hv);
      }
      *(v8bf*)(&cat[mrow * CATS + DIM + ncol0]) = hb.v;
    }
    __syncthreads();

    // ---- deep transition stack ----
    for (int li = 0; li < NLAYER; ++li) {
      const unsigned short* wr_p = wtWr + (size_t)li * HID * HID;
      const unsigned short* wz_p = wtWz + (size_t)li * HID * HID;
      const unsigned short* wc_p = wtWc + (size_t)li * HID * HID;
      v8f t0 = {}, t1 = {}, t2 = {};
#pragma unroll 2
      for (int kt = 0; kt < 8; ++kt) {
        v16bf ah = lds_a_frag(cat, am, kt, DIM, lh);
        t0 = WMMA_BF16(ah, g_b_frag(wr_p, wave, kt, 8, lane), t0);
        t1 = WMMA_BF16(ah, g_b_frag(wz_p, wave, kt, 8, lane), t1);
        t2 = WMMA_BF16(ah, g_b_frag(wc_p, wave, kt, 8, lane), t2);
      }
      {
        const int n  = wave * 16 + am;
        const int mb = lh * 8;
#pragma unroll
        for (int i = 0; i < 8; ++i) {
          S0[(mb + i) * HID + n] = t0[i];
          S1[(mb + i) * HID + n] = t1[i];
          S2[(mb + i) * HID + n] = t2[i];
        }
      }
      __syncthreads();
      ln_sigmoid_row(S0, wave, lane, tg_r + li * HID, tb_r + li * HID);  // ri
      ln_sigmoid_row(S1, wave, lane, tg_z + li * HID, tb_z + li * HID);  // zi
      // ri/zi for (row, cols) were produced by this same thread; S2/hbuf
      // already barriered -> no barrier needed before combine.
      {
        PackBF hb;
#pragma unroll
        for (int j = 0; j < 8; ++j) {
          const int n   = ncol0 + j;
          const int idx = mrow * HID + n;
          const float ni = fast_tanh(S0[idx] * (S2[idx] + tb_c[li * HID + n]));
          const float zi = S1[idx];
          const float hv = (1.f - zi) * ni + zi * hbuf[idx];
          hbuf[idx] = hv;
          hb.s[j] = f2bf(hv);
        }
        *(v8bf*)(&cat[mrow * CATS + DIM + ncol0]) = hb.v;
      }
      __syncthreads();
    }

    // ---- packed-sequence mask: freeze finished rows, zero their output ----
    {
      const bool valid = (t < mylen);
      PackBF hb;
      float of[8];
#pragma unroll
      for (int j = 0; j < 8; ++j) {
        const int idx = mrow * HID + ncol0 + j;
        const float hf = valid ? hbuf[idx] : hprev[idx];
        hbuf[idx]  = hf;
        hprev[idx] = hf;
        hb.s[j] = f2bf(hf);
        of[j]   = valid ? hf : 0.f;
      }
      *(v8bf*)(&cat[mrow * CATS + DIM + ncol0]) = hb.v;
      float* op = &out[(((size_t)t * BATCH) + b0 + mrow) * HID + ncol0];
      *(float4*)(op)     = make_float4(of[0], of[1], of[2], of[3]);
      *(float4*)(op + 4) = make_float4(of[4], of[5], of[6], of[7]);
    }
    __syncthreads();
  }
}

// ------------------------------ host launch --------------------------------
extern "C" void kernel_launch(void* const* d_in, const int* in_sizes, int n_in,
                              void* d_out, int out_size, void* d_ws, size_t ws_size,
                              hipStream_t stream) {
  (void)in_sizes; (void)n_in; (void)out_size; (void)ws_size;

  const float* x       = (const float*)d_in[0];
  const int*   lengths = (const int*)d_in[1];
  unsigned short* ws   = (unsigned short*)d_ws;

  // bf16-element offsets inside d_ws (total 1,376,256 elems = 2.75 MB)
  const size_t OFF_Wr  = 0;
  const size_t OFF_Wz  = 131072;
  const size_t OFF_Wl  = 262144;
  const size_t OFF_Cx  = 393216;
  const size_t OFF_Ch  = 458752;
  const size_t OFF_Wlt = 524288;
  const size_t OFF_tWr = 589824;
  const size_t OFF_tWz = 851968;
  const size_t OFF_tWc = 1114112;

  auto sw = [&](const float* src, size_t dstOff, int K, int N) {
    const int total = K * N;
    swizzle_w<<<(total + 255) / 256, 256, 0, stream>>>(src, ws + dstOff, K, N);
  };
  sw((const float*)d_in[2],  OFF_Wr,  512, 256);   // Wr
  sw((const float*)d_in[5],  OFF_Wz,  512, 256);   // Wz
  sw((const float*)d_in[8],  OFF_Wl,  512, 256);   // Wl
  sw((const float*)d_in[12], OFF_Cx,  256, 256);   // Cx
  sw((const float*)d_in[13], OFF_Ch,  256, 256);   // Ch_w
  sw((const float*)d_in[11], OFF_Wlt, 256, 256);   // Wlt
  for (int i = 0; i < NLAYER; ++i) {
    sw((const float*)d_in[15] + (size_t)i * 65536, OFF_tWr + (size_t)i * 65536, 256, 256);
    sw((const float*)d_in[18] + (size_t)i * 65536, OFF_tWz + (size_t)i * 65536, 256, 256);
    sw((const float*)d_in[21] + (size_t)i * 65536, OFF_tWc + (size_t)i * 65536, 256, 256);
  }

  const size_t shmem = (size_t)ROWS * CATS * 2     // cat (bf16)
                     + (size_t)8 * ROWS * HID * 4; // h, hprev, r, z, l, S0..S2
  rnn_kernel<<<dim3(BATCH / ROWS), dim3(512), shmem, stream>>>(
      x, lengths,
      ws + OFF_Wr, ws + OFF_Wz, ws + OFF_Wl,
      ws + OFF_Cx, ws + OFF_Ch, ws + OFF_Wlt,
      ws + OFF_tWr, ws + OFF_tWz, ws + OFF_tWc,
      (const float*)d_in[3],  (const float*)d_in[4],    // gr, br
      (const float*)d_in[6],  (const float*)d_in[7],    // gz, bz
      (const float*)d_in[9],  (const float*)d_in[10],   // gl, bl
      (const float*)d_in[14],                           // Ch_b
      (const float*)d_in[16], (const float*)d_in[17],   // tgr, tbr
      (const float*)d_in[19], (const float*)d_in[20],   // tgz, tbz
      (const float*)d_in[22],                           // tbc
      (float*)d_out);
}
